// LightningAttention2_75222057222262
// MI455X (gfx1250) — matrix operand order repaired and checked by reading, not compiled, not run on hardware
//
#include <hip/hip_runtime.h>
#include <hip/hip_bf16.h>
#include <cstdint>
#include <cstddef>

// ---------------------------------------------------------------------------
// Types for gfx1250 WMMA (wave32): D(16x16 f32) = A(16x32 bf16) x B(32x16 bf16) + C
// ---------------------------------------------------------------------------
typedef __attribute__((ext_vector_type(16))) __bf16       v16bf;
typedef __attribute__((ext_vector_type(8)))  float        v8f;
typedef __attribute__((ext_vector_type(8)))  unsigned int v8u;
typedef unsigned short ushort_t;

static constexpr int   DIMc  = 2048;
static constexpr int   HEADSc= 16;
static constexpr int   HD    = 128;   // head dim
static constexpr int   CH    = 128;   // chunk
static constexpr int   Bc    = 4;
static constexpr int   Nc    = 2048;
static constexpr int   NTOK  = Bc * Nc;          // 8192
static constexpr int   QKVN  = 3 * DIMc;         // 6144
static constexpr float F32_EPS = 1.1920928955078125e-7f;

// ---------------- scalar bf16 helpers (storage-only, RNE) -------------------
static __device__ __forceinline__ float bf2f(unsigned us) {
    unsigned u = us << 16;
    return __builtin_bit_cast(float, u);
}
static __device__ __forceinline__ ushort_t f2bf(float f) {
    unsigned u = __builtin_bit_cast(unsigned, f);
    unsigned r = (u + 0x7fffu + ((u >> 16) & 1u)) >> 16;
    return (ushort_t)r;
}

// ---------------- WMMA fragment loaders ------------------------------------
// A matrix (16x32 bf16, M=m0..m0+15, K=k0..k0+31), row-major source, ld in elems.
// Contiguous K order: lanes 0-15 (h=0): K 0..15 ; lanes 16-31: K 16..31.
static __device__ __forceinline__ v16bf load_frag_a(const ushort_t* base, size_t ld,
                                                    int m0, int k0, int lane) {
    int m  = m0 + (lane & 15);
    int hl = lane >> 4;
    const uint4* p0 = reinterpret_cast<const uint4*>(base + (size_t)m * ld + k0 + 16 * hl);
    const uint4* p1 = reinterpret_cast<const uint4*>(base + (size_t)m * ld + k0 + 8 + 16 * hl);
    uint4 x = *p0, y = *p1;
    v8u r;
    r[0] = x.x; r[1] = x.y; r[2] = x.z; r[3] = x.w;
    r[4] = y.x; r[5] = y.y; r[6] = y.z; r[7] = y.w;
    return __builtin_bit_cast(v16bf, r);
}

// B matrix (32x16 bf16, K=k0..k0+31, N=n0..n0+15) where the source is stored
// as rows of N with K contiguous (i.e. B[k][n] = src[n][k]) -- matches W^T GEMM
// and all transposed LDS operands. One contiguous 32B load per lane.
static __device__ __forceinline__ v16bf load_frag_bt(const ushort_t* base, size_t ld,
                                                     int n0, int k0, int lane) {
    int n  = n0 + (lane & 15);
    int hl = lane >> 4;
    const uint4* p = reinterpret_cast<const uint4*>(base + (size_t)n * ld + k0 + 16 * hl);
    uint4 x = p[0], y = p[1];
    v8u r;
    r[0] = x.x; r[1] = x.y; r[2] = x.z; r[3] = x.w;
    r[4] = y.x; r[5] = y.y; r[6] = y.z; r[7] = y.w;
    return __builtin_bit_cast(v16bf, r);
}

// Same as load_frag_bt but source is f32 (the attention state S^T); convert.
static __device__ __forceinline__ v16bf load_frag_bt_f32(const float* base, size_t ld,
                                                         int n0, int k0, int lane) {
    int n  = n0 + (lane & 15);
    int hl = lane >> 4;
    const float* p = base + (size_t)n * ld + k0 + 16 * hl;
    v8u r;
#pragma unroll
    for (int j = 0; j < 8; ++j) {
        unsigned lo = f2bf(p[2 * j]);
        unsigned hi = f2bf(p[2 * j + 1]);
        r[j] = lo | (hi << 16);
    }
    return __builtin_bit_cast(v16bf, r);
}

// Multiply every element of an A fragment by a scalar (used to fold lam_t).
static __device__ __forceinline__ v16bf scale_frag(v16bf a, float f) {
    v8u u = __builtin_bit_cast(v8u, a);
    v8u r;
#pragma unroll
    for (int i = 0; i < 8; ++i) {
        float f0 = bf2f(u[i] & 0xffffu) * f;
        float f1 = bf2f(u[i] >> 16) * f;
        r[i] = (unsigned)f2bf(f0) | ((unsigned)f2bf(f1) << 16);
    }
    return __builtin_bit_cast(v16bf, r);
}

static __device__ __forceinline__ v8f wmma_bf16(v16bf a, v16bf b, v8f c) {
    return __builtin_amdgcn_wmma_f32_16x16x32_bf16(false, a, false, b, (short)0, c,
                                                   false, false);
}

// ---------------------------------------------------------------------------
// 1) abs-mean -> scale (deterministic single-block reduction)
// ---------------------------------------------------------------------------
__global__ void absmean_scale_kernel(const float* __restrict__ w, long long n,
                                     float* __restrict__ scaleOut) {
    __shared__ float red[1024];
    float acc = 0.f;
    for (long long i = threadIdx.x; i < n; i += 1024)
        acc += fabsf(w[i]);
    red[threadIdx.x] = acc;
    __syncthreads();
    for (int s = 512; s > 0; s >>= 1) {
        if ((int)threadIdx.x < s) red[threadIdx.x] += red[threadIdx.x + s];
        __syncthreads();
    }
    if (threadIdx.x == 0) {
        float mean = red[0] / (float)n;
        scaleOut[0] = 1.0f / fmaxf(mean, 1e-5f);
    }
}

// 2) ternary quantize weights -> bf16
__global__ void quant_kernel(const float* __restrict__ w, ushort_t* __restrict__ wq,
                             long long n, const float* __restrict__ scalePtr) {
    float sc  = scalePtr[0];
    float inv = 1.0f / sc;
    long long stride = (long long)gridDim.x * blockDim.x;
    for (long long i = (long long)blockIdx.x * blockDim.x + threadIdx.x; i < n; i += stride) {
        float qv = fminf(1.f, fmaxf(-1.f, rintf(w[i] * sc))) * inv;
        wq[i] = f2bf(qv);
    }
}

// 3) RMSNorm fp32 row(2048) -> bf16
__global__ void rmsnorm_kernel(const float* __restrict__ x, const float* __restrict__ wgt,
                               ushort_t* __restrict__ out, float eps) {
    __shared__ float red[256];
    size_t row = blockIdx.x;
    int tid = threadIdx.x;
    float vals[8];
    float ss = 0.f;
#pragma unroll
    for (int j = 0; j < 8; ++j) {
        int col = j * 256 + tid;
        vals[j] = x[row * 2048 + col];
        ss += vals[j] * vals[j];
    }
    red[tid] = ss;
    __syncthreads();
    for (int s = 128; s > 0; s >>= 1) {
        if (tid < s) red[tid] += red[tid + s];
        __syncthreads();
    }
    float r = rsqrtf(red[0] * (1.0f / 2048.0f) + eps);
#pragma unroll
    for (int j = 0; j < 8; ++j) {
        int col = j * 256 + tid;
        out[row * 2048 + col] = f2bf(vals[j] * r * wgt[col]);
    }
}

// 4) double RMSNorm (norm_w, eps=1e-6) then (proj_norm_w, eps=f32eps) -> bf16
__global__ void rmsnorm2_kernel(const float* __restrict__ x, const float* __restrict__ w1,
                                const float* __restrict__ w2, ushort_t* __restrict__ out) {
    __shared__ float red[256];
    size_t row = blockIdx.x;
    int tid = threadIdx.x;
    float vals[8];
    float ss = 0.f;
#pragma unroll
    for (int j = 0; j < 8; ++j) {
        int col = j * 256 + tid;
        vals[j] = x[row * 2048 + col];
        ss += vals[j] * vals[j];
    }
    red[tid] = ss;
    __syncthreads();
    for (int s = 128; s > 0; s >>= 1) {
        if (tid < s) red[tid] += red[tid + s];
        __syncthreads();
    }
    float r1 = rsqrtf(red[0] * (1.0f / 2048.0f) + 1e-6f);
    __syncthreads();
    float ss2 = 0.f;
#pragma unroll
    for (int j = 0; j < 8; ++j) {
        int col = j * 256 + tid;
        vals[j] = vals[j] * r1 * w1[col];
        ss2 += vals[j] * vals[j];
    }
    red[tid] = ss2;
    __syncthreads();
    for (int s = 128; s > 0; s >>= 1) {
        if (tid < s) red[tid] += red[tid + s];
        __syncthreads();
    }
    float r2 = rsqrtf(red[0] * (1.0f / 2048.0f) + F32_EPS);
#pragma unroll
    for (int j = 0; j < 8; ++j) {
        int col = j * 256 + tid;
        out[row * 2048 + col] = f2bf(vals[j] * r2 * w2[col]);
    }
}

// ---------------------------------------------------------------------------
// 5) QKV GEMM: C[m,n] = sum_k A[m,k] * W[n,k]   (A: 8192x2048 bf16, W: 6144x2048 bf16)
//    128x128 block tiles, 8 waves, wave = 2x4 grid of 16x16 WMMA tiles.
//    Output scattered (bf16) to q/k/v in [B][N][H][D] = token-major x 2048 ch.
// ---------------------------------------------------------------------------
__global__ void gemm_qkv_kernel(const ushort_t* __restrict__ A, const ushort_t* __restrict__ W,
                                ushort_t* __restrict__ qb, ushort_t* __restrict__ kb,
                                ushort_t* __restrict__ vb) {
    int tid  = threadIdx.x;
    int w    = tid >> 5;
    int lane = tid & 31;
    int m0 = blockIdx.x * 128 + (w & 3) * 32;   // 2 m-tiles of 16
    int n0 = blockIdx.y * 128 + (w >> 2) * 64;  // 4 n-tiles of 16

    v8f acc[2][4];
#pragma unroll
    for (int i = 0; i < 2; ++i)
#pragma unroll
        for (int j = 0; j < 4; ++j) acc[i][j] = {};

    for (int k0 = 0; k0 < DIMc; k0 += 32) {
        v16bf a0 = load_frag_a(A, DIMc, m0,      k0, lane);
        v16bf a1 = load_frag_a(A, DIMc, m0 + 16, k0, lane);
#pragma unroll
        for (int j = 0; j < 4; ++j) {
            v16bf b = load_frag_bt(W, DIMc, n0 + 16 * j, k0, lane);
            acc[0][j] = wmma_bf16(a0, b, acc[0][j]);
            acc[1][j] = wmma_bf16(a1, b, acc[1][j]);
        }
    }

    int hl = lane >> 4, ln = lane & 15;
#pragma unroll
    for (int i = 0; i < 2; ++i)
#pragma unroll
        for (int j = 0; j < 4; ++j)
#pragma unroll
            for (int r = 0; r < 8; ++r) {
                int m  = m0 + 16 * i + r + 8 * hl;     // token index
                int gn = n0 + 16 * j + ln;             // 0..6143
                int third = gn >> 11;
                int ch    = gn & 2047;                 // h*128+d
                ushort_t bv = f2bf(acc[i][j][r]);
                ushort_t* dst = (third == 0) ? qb : (third == 1) ? kb : vb;
                dst[(size_t)m * 2048 + ch] = bv;
            }
}

// ---------------------------------------------------------------------------
// 6) Chunked lightning attention. One workgroup per (b,h). 8 waves.
//    LDS: qrow, krow, kT*lam_rev, vT, maskedScores (bf16, 32KB each)
//         + S^T state (128x128 f32, 64KB) = 224KB dynamic LDS.
// ---------------------------------------------------------------------------
__global__ void attn_kernel(const ushort_t* __restrict__ qg, const ushort_t* __restrict__ kg,
                            const ushort_t* __restrict__ vg, const float* __restrict__ s,
                            float* __restrict__ out) {
    extern __shared__ char smem[];
    ushort_t* qrow = (ushort_t*)smem;          // [t][d]  16384
    ushort_t* krow = qrow + 16384;             // [t][d]
    ushort_t* kTs  = krow + 16384;             // [d][t] * lam_rev[t]
    ushort_t* vT   = kTs + 16384;              // [e][t]
    ushort_t* scM  = vT + 16384;               // [t][s] masked scores bf16
    float*    ST   = (float*)(scM + 16384);    // [e][d] state transposed, f32

    int b = blockIdx.x >> 4;
    int h = blockIdx.x & 15;
    float sh   = s[h];
    float lamT = __expf(-sh * (float)CH);
    const float oscale = 0.088388347648318447f; // 1/sqrt(128)

    int tid  = threadIdx.x;
    int w    = tid >> 5;
    int lane = tid & 31;
    int hl   = lane >> 4, ln = lane & 15;

    for (int i = tid; i < 16384; i += 256) ST[i] = 0.f;
    __syncthreads();

    for (int c = 0; c < Nc / CH; ++c) {
        // ---- stage chunk into LDS -------------------------------------
        {
            int t    = tid >> 1;
            int d0   = (tid & 1) * 64;
            size_t gbase = (((size_t)b * Nc + (size_t)c * CH + t) * 2048) + (size_t)h * HD + d0;
            // q: straight vector copy
            const uint4* src = reinterpret_cast<const uint4*>(qg + gbase);
            uint4*       dst = reinterpret_cast<uint4*>(qrow + t * HD + d0);
#pragma unroll
            for (int j = 0; j < 8; ++j) dst[j] = src[j];
            // k: row copy + transposed copy scaled by lam_rev[t]
            float lrev = __expf(-sh * (float)(CH - t));
#pragma unroll
            for (int j = 0; j < 64; ++j) {
                ushort_t us = kg[gbase + j];
                krow[t * HD + d0 + j] = us;
                kTs[(d0 + j) * HD + t] = f2bf(bf2f(us) * lrev);
            }
            // v: transposed copy
#pragma unroll
            for (int j = 0; j < 64; ++j)
                vT[(d0 + j) * HD + t] = vg[gbase + j];
        }
        __syncthreads();

        // ---- scores = q k^T, masked, -> scM (bf16) --------------------
        {
            v8f sacc[8];
#pragma unroll
            for (int j = 0; j < 8; ++j) sacc[j] = {};
            for (int k0 = 0; k0 < HD; k0 += 32) {
                v16bf a = load_frag_a(qrow, HD, 16 * w, k0, lane);
#pragma unroll
                for (int tc = 0; tc < 8; ++tc) {
                    v16bf bb = load_frag_bt(krow, HD, 16 * tc, k0, lane);
                    sacc[tc] = wmma_bf16(a, bb, sacc[tc]);
                }
            }
#pragma unroll
            for (int tc = 0; tc < 8; ++tc)
#pragma unroll
                for (int r = 0; r < 8; ++r) {
                    int t    = 16 * w + r + 8 * hl;
                    int scol = 16 * tc + ln;
                    float m  = (t >= scol) ? __expf(-sh * (float)(t - scol)) : 0.f;
                    scM[t * CH + scol] = f2bf(sacc[tc][r] * m);
                }
        }
        __syncthreads();

        // ---- o = scale * ( (q*lam_t) @ S + scores @ v ) ---------------
        {
            v8f oacc[8];
#pragma unroll
            for (int j = 0; j < 8; ++j) oacc[j] = {};
            float lamt = __expf(-sh * (float)(16 * w + ln));  // A-lane row t
            for (int k0 = 0; k0 < HD; k0 += 32) {             // K over d
                v16bf a = scale_frag(load_frag_a(qrow, HD, 16 * w, k0, lane), lamt);
#pragma unroll
                for (int tc = 0; tc < 8; ++tc) {
                    v16bf bb = load_frag_bt_f32(ST, HD, 16 * tc, k0, lane);
                    oacc[tc] = wmma_bf16(a, bb, oacc[tc]);
                }
            }
            for (int k0 = 0; k0 < CH; k0 += 32) {             // K over s
                v16bf a = load_frag_a(scM, CH, 16 * w, k0, lane);
#pragma unroll
                for (int tc = 0; tc < 8; ++tc) {
                    v16bf bb = load_frag_bt(vT, CH, 16 * tc, k0, lane);
                    oacc[tc] = wmma_bf16(a, bb, oacc[tc]);
                }
            }
#pragma unroll
            for (int tc = 0; tc < 8; ++tc)
#pragma unroll
                for (int r = 0; r < 8; ++r) {
                    int t = 16 * w + r + 8 * hl;
                    int e = 16 * tc + ln;
                    size_t ntok = (size_t)c * CH + t;
                    out[((size_t)b * Nc + ntok) * DIMc + (size_t)h * HD + e] =
                        oscale * oacc[tc][r];
                }
        }
        __syncthreads();  // everyone done reading old S

        // ---- kv = (k^T lam_rev) @ v ; S = lamT*S + kv -----------------
        {
            v8f kacc[8];
#pragma unroll
            for (int j = 0; j < 8; ++j) kacc[j] = {};
            for (int k0 = 0; k0 < CH; k0 += 32) {             // K over t
                v16bf a = load_frag_a(kTs, HD, 16 * w, k0, lane);
#pragma unroll
                for (int tc = 0; tc < 8; ++tc) {
                    v16bf bb = load_frag_bt(vT, CH, 16 * tc, k0, lane);
                    kacc[tc] = wmma_bf16(a, bb, kacc[tc]);
                }
            }
#pragma unroll
            for (int tc = 0; tc < 8; ++tc)
#pragma unroll
                for (int r = 0; r < 8; ++r) {
                    int d = 16 * w + r + 8 * hl;
                    int e = 16 * tc + ln;
                    int idx = e * HD + d;                      // S^T[e][d]
                    ST[idx] = lamT * ST[idx] + kacc[tc][r];
                }
        }
        __syncthreads();  // before next chunk overwrites staging LDS
    }
}

// ---------------------------------------------------------------------------
// 7) Proj GEMM: out[m,n] = sum_k A[m,k] * W[n,k] -> f32 d_out (8192x2048)
// ---------------------------------------------------------------------------
__global__ void gemm_proj_kernel(const ushort_t* __restrict__ A, const ushort_t* __restrict__ W,
                                 float* __restrict__ out) {
    int tid  = threadIdx.x;
    int w    = tid >> 5;
    int lane = tid & 31;
    int m0 = blockIdx.x * 128 + (w & 3) * 32;
    int n0 = blockIdx.y * 128 + (w >> 2) * 64;

    v8f acc[2][4];
#pragma unroll
    for (int i = 0; i < 2; ++i)
#pragma unroll
        for (int j = 0; j < 4; ++j) acc[i][j] = {};

    for (int k0 = 0; k0 < DIMc; k0 += 32) {
        v16bf a0 = load_frag_a(A, DIMc, m0,      k0, lane);
        v16bf a1 = load_frag_a(A, DIMc, m0 + 16, k0, lane);
#pragma unroll
        for (int j = 0; j < 4; ++j) {
            v16bf b = load_frag_bt(W, DIMc, n0 + 16 * j, k0, lane);
            acc[0][j] = wmma_bf16(a0, b, acc[0][j]);
            acc[1][j] = wmma_bf16(a1, b, acc[1][j]);
        }
    }

    int hl = lane >> 4, ln = lane & 15;
#pragma unroll
    for (int i = 0; i < 2; ++i)
#pragma unroll
        for (int j = 0; j < 4; ++j)
#pragma unroll
            for (int r = 0; r < 8; ++r) {
                int m  = m0 + 16 * i + r + 8 * hl;
                int gn = n0 + 16 * j + ln;
                out[(size_t)m * DIMc + gn] = acc[i][j][r];
            }
}

// ---------------------------------------------------------------------------
// Host launcher
// ---------------------------------------------------------------------------
extern "C" void kernel_launch(void* const* d_in, const int* in_sizes, int n_in,
                              void* d_out, int out_size, void* d_ws, size_t ws_size,
                              hipStream_t stream) {
    (void)in_sizes; (void)n_in; (void)out_size; (void)ws_size;

    const float* x           = (const float*)d_in[0];
    const float* qkv_w       = (const float*)d_in[1];
    const float* qkv_norm_w  = (const float*)d_in[2];
    const float* proj_w      = (const float*)d_in[3];
    const float* proj_norm_w = (const float*)d_in[4];
    const float* norm_w      = (const float*)d_in[5];
    const float* s_dec       = (const float*)d_in[6];
    float* outp              = (float*)d_out;

    char* ws = (char*)d_ws;
    size_t off = 0;
    float* scale0 = (float*)(ws + off);
    float* scale1 = scale0 + 1;
    off += 256;
    ushort_t* wq_qkv  = (ushort_t*)(ws + off); off += (size_t)QKVN * DIMc * 2;  // 24 MB
    ushort_t* wq_proj = (ushort_t*)(ws + off); off += (size_t)DIMc * DIMc * 2;  // 8 MB
    ushort_t* hbuf    = (ushort_t*)(ws + off); off += (size_t)NTOK * DIMc * 2;  // 32 MB
    ushort_t* qb      = (ushort_t*)(ws + off); off += (size_t)NTOK * DIMc * 2;
    ushort_t* kb      = (ushort_t*)(ws + off); off += (size_t)NTOK * DIMc * 2;
    ushort_t* vb      = (ushort_t*)(ws + off); off += (size_t)NTOK * DIMc * 2;
    float*    attno   = (float*)(ws + off);    off += (size_t)NTOK * DIMc * 4;  // 64 MB
    ushort_t* y2      = (ushort_t*)(ws + off); off += (size_t)NTOK * DIMc * 2;

    const long long nQkvW  = (long long)QKVN * DIMc;   // 12,582,912
    const long long nProjW = (long long)DIMc * DIMc;   //  4,194,304

    absmean_scale_kernel<<<1, 1024, 0, stream>>>(qkv_w,  nQkvW,  scale0);
    absmean_scale_kernel<<<1, 1024, 0, stream>>>(proj_w, nProjW, scale1);
    quant_kernel<<<(unsigned)((nQkvW  + 1023) / 1024), 1024, 0, stream>>>(qkv_w,  wq_qkv,  nQkvW,  scale0);
    quant_kernel<<<(unsigned)((nProjW + 1023) / 1024), 1024, 0, stream>>>(proj_w, wq_proj, nProjW, scale1);

    rmsnorm_kernel<<<NTOK, 256, 0, stream>>>(x, qkv_norm_w, hbuf, F32_EPS);

    gemm_qkv_kernel<<<dim3(NTOK / 128, QKVN / 128), 256, 0, stream>>>(hbuf, wq_qkv, qb, kb, vb);

    constexpr size_t attnLds = 5u * 32768u + 65536u;  // 229,376 B < 320 KB/WGP
    attn_kernel<<<Bc * HEADSc, 256, attnLds, stream>>>(qb, kb, vb, s_dec, attno);

    rmsnorm2_kernel<<<NTOK, 256, 0, stream>>>(attno, norm_w, proj_norm_w, y2);

    gemm_proj_kernel<<<dim3(NTOK / 128, DIMc / 128), 256, 0, stream>>>(y2, wq_proj, outp);
}
